// Attention_15479062135535
// MI455X (gfx1250) — matrix operand order repaired and checked by reading, not compiled, not run on hardware
//
#include <hip/hip_runtime.h>
#include <cstdint>

// Problem constants (from reference)
#define B_     4
#define N_     2048
#define DIM_   512
#define HEADS_ 8
#define DH_    64
#define HID_   512
#define THID_  1536   // 3*HID

typedef __attribute__((ext_vector_type(16))) __bf16 v16bf;
typedef __attribute__((ext_vector_type(8)))  float  v8f;
typedef unsigned short u16;
typedef unsigned int   u32;

// exact pointer types for the gfx1250 async-to-LDS builtin
// (param0 is 'int __vector(4) __device__*', i.e. v4i in AS1; LDS side is AS3)
typedef int vi4 __attribute__((vector_size(16)));
typedef __attribute__((address_space(1))) vi4 g_vi4;
typedef __attribute__((address_space(3))) vi4 l_vi4;

union FragH {
  v16bf v;
  u32   u[8];
  uint4 q4[2];
};

__device__ __forceinline__ u16 f2bf(float f) {
  union { __bf16 h; u16 s; } cv;
  cv.h = (__bf16)f;            // native v_cvt (RNE), 1 VALU op
  return cv.s;
}

__device__ __forceinline__ v8f wmma_bf16(const FragH& a, const FragH& b, v8f c) {
  // (neg_a, A, neg_b, B, c_mod, C, reuse_a, reuse_b)
  return __builtin_amdgcn_wmma_f32_16x16x32_bf16(false, a.v, false, b.v,
                                                 (short)0, c, false, false);
}

// xor-butterfly across a 16-lane row via DPP ROW_XMASK (no LDS traffic)
template<int MASK>
__device__ __forceinline__ float xor16f(float x) {
#if __has_builtin(__builtin_amdgcn_update_dpp)
  return __int_as_float(__builtin_amdgcn_update_dpp(
      0, __float_as_int(x), 0x160 | MASK, 0xf, 0xf, true));
#else
  return __shfl_xor(x, MASK, 32);
#endif
}

__device__ __forceinline__ float row16_max(float x) {
  x = fmaxf(x, xor16f<1>(x));
  x = fmaxf(x, xor16f<2>(x));
  x = fmaxf(x, xor16f<4>(x));
  x = fmaxf(x, xor16f<8>(x));
  return x;
}

__device__ __forceinline__ float row16_sum(float x) {
  x += xor16f<1>(x);
  x += xor16f<2>(x);
  x += xor16f<4>(x);
  x += xor16f<8>(x);
  return x;
}

// 16B global -> LDS copy; async (ASYNCcnt) when the gfx1250 builtin exists
__device__ __forceinline__ void cp_b128(const u16* g, u16* l) {
#if __has_builtin(__builtin_amdgcn_global_load_async_to_lds_b128)
  __builtin_amdgcn_global_load_async_to_lds_b128(
      (g_vi4*)(uintptr_t)g, (l_vi4*)(uintptr_t)l, 0, 0);
#else
  *(uint4*)l = *(const uint4*)g;
#endif
}

__device__ __forceinline__ void wait_async() {
#if __has_builtin(__builtin_amdgcn_s_wait_asynccnt)
  __builtin_amdgcn_s_wait_asynccnt(0);
#elif __has_builtin(__builtin_amdgcn_global_load_async_to_lds_b128)
  asm volatile("s_wait_asynccnt 0x0" ::: "memory");
#endif
}

// ---------------------------------------------------------------------------
// Tiled GEMM: C[M,NC] = A[M,K] @ W[K,NC]; A fp32 or bf16, W fp32.
// Block tile 64x64, 4 waves (wave w -> rows 16w..16w+15), K step 32.
// OUT_BF16: write bf16, scaling columns n < scale_ncols by `scale`.
// ---------------------------------------------------------------------------
template<bool A_BF16, bool OUT_BF16>
__global__ __launch_bounds__(256)
void gemm_wmma(const void* __restrict__ Ap, const float* __restrict__ W,
               void* __restrict__ Cp, int M, int K, int NC,
               float scale, int scale_ncols)
{
  __shared__ alignas(16) u16 Asl[64 * 32];   // [m_local][k]  (row-major, stride 32)
  __shared__ alignas(16) u16 Bsl[64 * 32];   // [n_local][k]  (transposed W tile)

  const int tid  = threadIdx.x;
  const int wid  = tid >> 5;
  const int lane = tid & 31;
  const int half = lane >> 4;
  const int lr   = lane & 15;

  const int bn0 = blockIdx.x * 64;
  const int bm0 = blockIdx.y * 64;

  const v8f vzero = {0.f,0.f,0.f,0.f,0.f,0.f,0.f,0.f};
  v8f acc[4];
#pragma unroll
  for (int i = 0; i < 4; ++i) acc[i] = vzero;

  const int arow = tid >> 2;        // 0..63
  const int ac0  = (tid & 3) * 8;   // 0,8,16,24
  const int wkl  = tid >> 3;        // 0..31
  const int wn0  = (tid & 7) * 8;   // 0..56

  for (int k0 = 0; k0 < K; k0 += 32) {
    __syncthreads();
    // ---- stage A tile (64x32) as bf16 ----
    if constexpr (A_BF16) {
      const u16* Ab = (const u16*)Ap;
      cp_b128(Ab + (size_t)(bm0 + arow) * K + k0 + ac0, &Asl[arow * 32 + ac0]);
    } else {
      const float* Af = (const float*)Ap + (size_t)(bm0 + arow) * K + k0 + ac0;
      float4 f0 = *(const float4*)Af;
      float4 f1 = *(const float4*)(Af + 4);
      uint4 d;
      d.x = (u32)f2bf(f0.x) | ((u32)f2bf(f0.y) << 16);
      d.y = (u32)f2bf(f0.z) | ((u32)f2bf(f0.w) << 16);
      d.z = (u32)f2bf(f1.x) | ((u32)f2bf(f1.y) << 16);
      d.w = (u32)f2bf(f1.z) | ((u32)f2bf(f1.w) << 16);
      *(uint4*)(&Asl[arow * 32 + ac0]) = d;
    }
    // ---- stage W tile (32x64), transposed into [n][k] ----
    {
      const float* Wf = W + (size_t)(k0 + wkl) * NC + bn0 + wn0;
      float4 f0 = *(const float4*)Wf;
      float4 f1 = *(const float4*)(Wf + 4);
      float v[8] = {f0.x,f0.y,f0.z,f0.w,f1.x,f1.y,f1.z,f1.w};
#pragma unroll
      for (int ii = 0; ii < 8; ++ii)
        Bsl[(wn0 + ii) * 32 + wkl] = f2bf(v[ii]);
    }
    if constexpr (A_BF16) wait_async();
    __syncthreads();

    // A fragment (16x32), k = half*16 + e (contiguous, same order as B)
    FragH a;
#pragma unroll
    for (int vp = 0; vp < 8; ++vp) {
      const int kk = half * 16 + vp * 2;
      a.u[vp] = *(const u32*)&Asl[(wid * 16 + lr) * 32 + kk];
    }
#pragma unroll
    for (int bn = 0; bn < 4; ++bn) {
      FragH b;   // B fragment: n = lane&15, k = half*16 + e -> contiguous in Bsl
      const uint4* p = (const uint4*)&Bsl[(bn * 16 + lr) * 32 + half * 16];
      b.q4[0] = p[0];
      b.q4[1] = p[1];
      acc[bn] = wmma_bf16(a, b, acc[bn]);
    }
  }

  // C layout: element r -> row r + 8*half, col = lane&15
#pragma unroll
  for (int bn = 0; bn < 4; ++bn) {
#pragma unroll
    for (int r = 0; r < 8; ++r) {
      const int m = bm0 + wid * 16 + r + 8 * half;
      const int n = bn0 + bn * 16 + lr;
      float val = acc[bn][r];
      if constexpr (OUT_BF16) {
        if (n < scale_ncols) val *= scale;
        ((u16*)Cp)[(size_t)m * NC + n] = f2bf(val);
      } else {
        ((float*)Cp)[(size_t)m * NC + n] = val;
      }
    }
  }
}

// ---------------------------------------------------------------------------
// Flash attention: one block per (b, h, 64-query tile); 4 waves x 16 rows.
// qkv is bf16 (q already scaled by DH^-0.5). Streams keys in steps of 32.
// ---------------------------------------------------------------------------
__global__ __launch_bounds__(256)
void attn_flash(const u16* __restrict__ qkv, const float* __restrict__ pb,
                const unsigned char* __restrict__ focus, u16* __restrict__ o)
{
  __shared__ alignas(16) u16 Klds[32 * 64];     // [j_local][d]
  __shared__ alignas(16) u16 Vlds[64 * 32];     // transposed: [d][j_local]
  __shared__ alignas(16) u16 Plds[4 * 16 * 32]; // per-wave P relayout slab

  const int tid  = threadIdx.x;
  const int wid  = tid >> 5;
  const int lane = tid & 31;
  const int half = lane >> 4;
  const int lr   = lane & 15;

  const int bh = blockIdx.y;
  const int b  = bh >> 3;
  const int h  = bh & 7;
  const int i0 = blockIdx.x * 64;

  if (focus[b]) {
    // mask == eye: softmax is an exact one-hot -> out rows = V rows
    const int r  = tid >> 2;
    const int c0 = (tid & 3) * 16;
    const size_t src = (size_t)(b * N_ + i0 + r) * THID_ + 2 * HID_ + h * DH_ + c0;
    const size_t dst = (size_t)(b * N_ + i0 + r) * HID_ + h * DH_ + c0;
    uint4 x0 = *(const uint4*)(qkv + src);
    uint4 x1 = *(const uint4*)(qkv + src + 8);
    *(uint4*)(o + dst)     = x0;
    *(uint4*)(o + dst + 8) = x1;
    return;
  }

  const int iw = i0 + wid * 16;

  // Q fragments for d=[0,32) and [32,64) (A layout, loaded once)
  FragH qf[2];
  {
    const size_t qbase = (size_t)(b * N_ + iw + lr) * THID_ + h * DH_;
#pragma unroll
    for (int dk = 0; dk < 2; ++dk)
#pragma unroll
      for (int vp = 0; vp < 8; ++vp) {
        const int kk = half * 16 + vp * 2;
        qf[dk].u[vp] = *(const u32*)(qkv + qbase + dk * 32 + kk);
      }
  }

  const v8f vzero = {0.f,0.f,0.f,0.f,0.f,0.f,0.f,0.f};
  v8f accO[4];
  float mrun[8], lrun[8];
#pragma unroll
  for (int c = 0; c < 4; ++c) accO[c] = vzero;
#pragma unroll
  for (int r = 0; r < 8; ++r) { mrun[r] = -__builtin_inff(); lrun[r] = 0.f; }

  const int ldj = tid >> 3;        // 0..31
  const int ldd = (tid & 7) * 8;   // 0..56

  for (int jb = 0; jb < N_ / 32; ++jb) {
    const int j0 = jb * 32;
    __syncthreads();
    {
      const size_t rbase = (size_t)(b * N_ + j0 + ldj) * THID_ + h * DH_ + ldd;
      // K tile: straight copy -> async global->LDS path
      cp_b128(qkv + rbase + HID_, &Klds[ldj * 64 + ldd]);
      // V tile: needs transpose, goes through VGPRs
      uint4 vd = *(const uint4*)(qkv + rbase + 2 * HID_);
      const u16* vv = (const u16*)&vd;
#pragma unroll
      for (int ii = 0; ii < 8; ++ii)
        Vlds[(ldd + ii) * 32 + ldj] = vv[ii];
    }
    wait_async();
    __syncthreads();

    // S tiles for key cols [j0, j0+16) and [j0+16, j0+32)
    v8f s[2];
    s[0] = vzero; s[1] = vzero;
#pragma unroll
    for (int t = 0; t < 2; ++t)
#pragma unroll
      for (int dk = 0; dk < 2; ++dk) {
        FragH kb;  // B frag: n = j (lane&15), k = d chunk, contiguous in Klds
        const uint4* p = (const uint4*)&Klds[(t * 16 + lr) * 64 + dk * 32 + half * 16];
        kb.q4[0] = p[0]; kb.q4[1] = p[1];
        s[t] = wmma_bf16(qf[dk], kb, s[t]);
      }

    // + pos_bias, online softmax (row stats redundant per 16-lane half)
#pragma unroll
    for (int r = 0; r < 8; ++r) {
      const int ig = iw + r + 8 * half;
      const size_t pbb = ((size_t)h * N_ + ig) * N_ + j0;
      float s0 = s[0][r] + pb[pbb + lr];
      float s1 = s[1][r] + pb[pbb + 16 + lr];
      if (jb + 1 < N_ / 32)
        __builtin_prefetch(&pb[pbb + 32 + lr], 0, 0);   // next bias block -> L2
      const float mx   = row16_max(fmaxf(s0, s1));
      const float mnew = fmaxf(mrun[r], mx);
      const float corr = __expf(mrun[r] - mnew);
      const float p0 = __expf(s0 - mnew);
      const float p1 = __expf(s1 - mnew);
      const float rs = row16_sum(p0 + p1);
      lrun[r] = lrun[r] * corr + rs;
      mrun[r] = mnew;
#pragma unroll
      for (int c = 0; c < 4; ++c) accO[c][r] *= corr;
      // stash P (C layout coords) for relayout to A layout
      Plds[wid * 512 + (r + 8 * half) * 32 + lr]      = f2bf(p0);
      Plds[wid * 512 + (r + 8 * half) * 32 + 16 + lr] = f2bf(p1);
    }

    // same-wave LDS store->load: LDS is in-order per wave; fence + wait anyway
    __builtin_amdgcn_wave_barrier();
    asm volatile("s_wait_dscnt 0x0" ::: "memory");
    __builtin_amdgcn_wave_barrier();

    FragH pf;  // P as A fragment (16 rows x 32 j)
#pragma unroll
    for (int vp = 0; vp < 8; ++vp) {
      const int kk = half * 16 + vp * 2;
      pf.u[vp] = *(const u32*)&Plds[wid * 512 + lr * 32 + kk];
    }
#pragma unroll
    for (int c = 0; c < 4; ++c) {
      FragH vf;  // B frag: n = d (lane&15), k = j, contiguous in transposed Vlds
      const uint4* p = (const uint4*)&Vlds[(c * 16 + lr) * 32 + half * 16];
      vf.q4[0] = p[0]; vf.q4[1] = p[1];
      accO[c] = wmma_bf16(pf, vf, accO[c]);
    }
  }

  // normalize and write O (bf16, head-interleaved (B,N,HID) layout)
#pragma unroll
  for (int r = 0; r < 8; ++r) {
    const float inv = 1.0f / lrun[r];
    const int ig = iw + r + 8 * half;
    const size_t ob = (size_t)(b * N_ + ig) * HID_ + h * DH_;
#pragma unroll
    for (int c = 0; c < 4; ++c)
      o[ob + c * 16 + lr] = f2bf(accO[c][r] * inv);
  }
}

// ---------------------------------------------------------------------------
extern "C" void kernel_launch(void* const* d_in, const int* in_sizes, int n_in,
                              void* d_out, int out_size, void* d_ws, size_t ws_size,
                              hipStream_t stream)
{
  (void)in_sizes; (void)n_in; (void)out_size; (void)ws_size;
  const float* x    = (const float*)d_in[0];                 // (B,N,DIM)
  const float* pb   = (const float*)d_in[1];                 // (H,N,N)
  const unsigned char* fm = (const unsigned char*)d_in[2];   // (B,) bool
  const float* wqkv = (const float*)d_in[3];                 // (DIM, 3*HID)
  const float* wout = (const float*)d_in[4];                 // (HID, DIM)
  float* out = (float*)d_out;                                // (B,N,DIM) f32

  u16* qkv  = (u16*)d_ws;                                    // bf16 (B*N, 3*HID)
  u16* attn = qkv + (size_t)B_ * N_ * THID_;                 // bf16 (B*N, HID)

  dim3 blk(256);
  // QKV projection (q columns pre-scaled by DH^-0.5 = 0.125)
  gemm_wmma<false, true><<<dim3(THID_ / 64, (B_ * N_) / 64), blk, 0, stream>>>(
      x, wqkv, qkv, B_ * N_, DIM_, THID_, 0.125f, HID_);
  // Flash attention
  attn_flash<<<dim3(N_ / 64, B_ * HEADS_), blk, 0, stream>>>(qkv, pb, fm, attn);
  // Output projection -> fp32
  gemm_wmma<true, false><<<dim3(HID_ / 64, (B_ * N_) / 64), blk, 0, stream>>>(
      attn, wout, out, B_ * N_, HID_, DIM_, 1.0f, 0);
}
